// TorchLMHeadJSD_56092272886458
// MI455X (gfx1250) — matrix operand order repaired and checked by reading, not compiled, not run on hardware
//
#include <hip/hip_runtime.h>

typedef __attribute__((ext_vector_type(16))) __bf16 v16bf;
typedef __attribute__((ext_vector_type(8)))  __bf16 v8bf;
typedef __attribute__((ext_vector_type(4)))  __bf16 v4bf;
typedef __attribute__((ext_vector_type(8)))  float  v8f;
typedef __attribute__((ext_vector_type(4)))  int    vi4;

// gfx1250 async global->LDS path (ASYNCcnt), guarded so absence of the
// builtin falls back to the proven VGPR-staging path.
#if defined(__has_builtin)
#  if __has_builtin(__builtin_amdgcn_global_load_async_to_lds_b128)
#    define USE_ASYNC_LDS 1
#  endif
#endif
#ifndef USE_ASYNC_LDS
#  define USE_ASYNC_LDS 0
#endif

#if defined(__has_builtin) && __has_builtin(__builtin_amdgcn_s_wait_asynccnt)
#  define WAIT_ASYNC() __builtin_amdgcn_s_wait_asynccnt(0)
#else
#  define WAIT_ASYNC() asm volatile("s_wait_asynccnt 0x0" ::: "memory")
#endif

// Builtin signature (from hipcc diagnostic): (int4 AS1*, int4 AS3*, imm, imm)
typedef __attribute__((address_space(1))) vi4* gvi4p;   // global int4*
typedef __attribute__((address_space(3))) vi4* lvi4p;   // LDS int4*

// Problem geometry (fixed by the reference).
constexpr int BT_  = 2048;          // tokens
constexpr int VDIM = 32000;         // vocab
constexpr int HS_  = 2048;          // student hidden
constexpr int HT_  = 4096;          // teacher hidden
constexpr int MT   = 64;            // rows per workgroup tile
constexpr int NT   = 128;           // vocab cols per workgroup tile
constexpr int KC   = 32;            // K chunk staged in LDS (one WMMA k-step)
constexpr int NCH  = VDIM / NT;     // 250 vocab chunks
constexpr int NRT  = BT_ / MT;      // 32 row tiles

#if USE_ASYNC_LDS
typedef float  stage_t;             // raw fp32 staged by the async engine
constexpr int KCP = KC + 4;         // +16B row pad: stride 36 dwords, gcd(36,64)=4
#else
typedef __bf16 stage_t;             // bf16 staged through VGPRs
constexpr int KCP = KC + 8;         // +16B row pad: stride 20 dwords, gcd(20,64)=4
#endif
constexpr int AE = MT * KCP;        // A staging elems per buffer
constexpr int BE = NT * KCP;        // B staging elems per buffer

__device__ __forceinline__ v16bf cat8(v8bf lo, v8bf hi) {
  return __builtin_shufflevector(lo, hi, 0,1,2,3,4,5,6,7,8,9,10,11,12,13,14,15);
}

// Build a bf16 fragment from two 8-float spans in LDS (async path).
__device__ __forceinline__ v16bf frag16_f32(const float* p0, const float* p1) {
  float4 a = *(const float4*)(p0 + 0);
  float4 b = *(const float4*)(p0 + 4);
  float4 c = *(const float4*)(p1 + 0);
  float4 d = *(const float4*)(p1 + 4);
  v16bf r;
  r[0]  = (__bf16)a.x; r[1]  = (__bf16)a.y; r[2]  = (__bf16)a.z; r[3]  = (__bf16)a.w;
  r[4]  = (__bf16)b.x; r[5]  = (__bf16)b.y; r[6]  = (__bf16)b.z; r[7]  = (__bf16)b.w;
  r[8]  = (__bf16)c.x; r[9]  = (__bf16)c.y; r[10] = (__bf16)c.z; r[11] = (__bf16)c.w;
  r[12] = (__bf16)d.x; r[13] = (__bf16)d.y; r[14] = (__bf16)d.z; r[15] = (__bf16)d.w;
  return r;
}

// 64x128 tile of Z = X[row0:+64,:] * W[col0:+128,:]^T, accumulated in
// registers. 256 threads = 8 waves in a 2x4 grid; wave (wm,wn) owns rows
// wm*32..+32 and cols wn*32..+32 -> acc[2][2] (4 WMMAs per k-step).
// Double-buffered LDS staging (bank-conflict-free padded rows), one barrier
// per K chunk.
__device__ __forceinline__ void gemm64(
    const float* __restrict__ X, const float* __restrict__ W, const int HDIM,
    const int row0, const int col0, stage_t* lA, stage_t* lB,
    const int tid, v8f acc[2][2])
{
  const int lane = tid & 31;
  const int wave = tid >> 5;
  const int wm = wave >> 2;      // 0..1
  const int wn = wave & 3;       // 0..3
  const int m  = lane & 15;
  const int half = lane >> 4;

  {
    v8f z = {};
    #pragma unroll
    for (int r = 0; r < 2; ++r)
      #pragma unroll
      for (int c = 0; c < 2; ++c) acc[r][c] = z;
  }

  // Fixed per-thread staging slots (float4 granularity, 8 float4 per row).
  // A: 64x32 fp32 = 512 float4 (2/thread); B: 128x32 = 1024 float4 (4/thread).
  int ar[2], ac[2], br[4], bc[4];
  #pragma unroll
  for (int i = 0; i < 2; ++i) { int x = tid + i * 256; ar[i] = x >> 3; ac[i] = (x & 7) * 4; }
  #pragma unroll
  for (int i = 0; i < 4; ++i) { int x = tid + i * 256; br[i] = x >> 3; bc[i] = (x & 7) * 4; }

#if USE_ASYNC_LDS
  auto stage = [&](int k0, stage_t* dA, stage_t* dB) {
    #pragma unroll
    for (int i = 0; i < 2; ++i)
      __builtin_amdgcn_global_load_async_to_lds_b128(
          (gvi4p)(X + (size_t)(row0 + ar[i]) * HDIM + k0 + ac[i]),
          (lvi4p)&dA[ar[i] * KCP + ac[i]], 0, 0);
    #pragma unroll
    for (int i = 0; i < 4; ++i)
      __builtin_amdgcn_global_load_async_to_lds_b128(
          (gvi4p)(W + (size_t)(col0 + br[i]) * HDIM + k0 + bc[i]),
          (lvi4p)&dB[br[i] * KCP + bc[i]], 0, 0);
  };
#else
  float4 ra[2], rb[4];
  auto loadg = [&](int k0) {
    #pragma unroll
    for (int i = 0; i < 2; ++i)
      ra[i] = *(const float4*)(X + (size_t)(row0 + ar[i]) * HDIM + k0 + ac[i]);
    #pragma unroll
    for (int i = 0; i < 4; ++i)
      rb[i] = *(const float4*)(W + (size_t)(col0 + br[i]) * HDIM + k0 + bc[i]);
  };
  auto stores = [&](stage_t* dA, stage_t* dB) {
    #pragma unroll
    for (int i = 0; i < 2; ++i) {
      v4bf p; p.x = (__bf16)ra[i].x; p.y = (__bf16)ra[i].y; p.z = (__bf16)ra[i].z; p.w = (__bf16)ra[i].w;
      *(v4bf*)&dA[ar[i] * KCP + ac[i]] = p;
    }
    #pragma unroll
    for (int i = 0; i < 4; ++i) {
      v4bf p; p.x = (__bf16)rb[i].x; p.y = (__bf16)rb[i].y; p.z = (__bf16)rb[i].z; p.w = (__bf16)rb[i].w;
      *(v4bf*)&dB[br[i] * KCP + bc[i]] = p;
    }
  };
#endif

  auto compute = [&](const stage_t* cA, const stage_t* cB) {
    v16bf af[2], bfr[2];
#if USE_ASYNC_LDS
    #pragma unroll
    for (int r = 0; r < 2; ++r) {
      const float* p = &cA[(wm * 32 + r * 16 + m) * KCP + half * 16];
      af[r] = frag16_f32(p, p + 8);
    }
    #pragma unroll
    for (int c = 0; c < 2; ++c) {
      const float* p = &cB[(wn * 32 + c * 16 + m) * KCP + half * 16];
      bfr[c] = frag16_f32(p, p + 8);
    }
#else
    #pragma unroll
    for (int r = 0; r < 2; ++r) {
      const __bf16* p = &cA[(wm * 32 + r * 16 + m) * KCP + half * 16];
      af[r] = cat8(*(const v8bf*)p, *(const v8bf*)(p + 8));
    }
    #pragma unroll
    for (int c = 0; c < 2; ++c) {
      const __bf16* p = &cB[(wn * 32 + c * 16 + m) * KCP + half * 16];
      bfr[c] = cat8(*(const v8bf*)p, *(const v8bf*)(p + 8));
    }
#endif
    #pragma unroll
    for (int r = 0; r < 2; ++r)
      #pragma unroll
      for (int c = 0; c < 2; ++c)
        acc[r][c] = __builtin_amdgcn_wmma_f32_16x16x32_bf16(
            false, af[r], false, bfr[c], (short)0, acc[r][c], false, false);
  };

  const int nck = HDIM / KC;
#if USE_ASYNC_LDS
  stage(0, lA, lB);
  WAIT_ASYNC();
  __syncthreads();
  for (int ck = 0; ck < nck; ++ck) {
    const int cur = ck & 1, nxt = cur ^ 1;
    if (ck + 1 < nck) stage((ck + 1) * KC, lA + nxt * AE, lB + nxt * BE);
    if (ck + 2 < nck) {
      __builtin_prefetch(W + (size_t)(col0 + br[0]) * HDIM + (ck + 2) * KC + bc[0], 0, 1);
      __builtin_prefetch(X + (size_t)(row0 + ar[0]) * HDIM + (ck + 2) * KC + ac[0], 0, 1);
    }
    compute(lA + cur * AE, lB + cur * BE);
    WAIT_ASYNC();
    __syncthreads();
  }
#else
  loadg(0);
  stores(lA, lB);
  __syncthreads();
  for (int ck = 0; ck < nck; ++ck) {
    const int cur = ck & 1, nxt = cur ^ 1;
    const bool more = (ck + 1) < nck;
    if (more) loadg((ck + 1) * KC);
    if (ck + 2 < nck) {
      __builtin_prefetch(W + (size_t)(col0 + br[0]) * HDIM + (ck + 2) * KC + bc[0], 0, 1);
      __builtin_prefetch(X + (size_t)(row0 + ar[0]) * HDIM + (ck + 2) * KC + ac[0], 0, 1);
    }
    compute(lA + cur * AE, lB + cur * BE);
    if (more) stores(lA + nxt * AE, lB + nxt * BE);
    __syncthreads();
  }
#endif
}

// Per-chunk (max, sum-exp) per row, straight from the accumulators.
// C/D layout: VGPR i, lanes 0-15 -> (M = base+i, N = lane); lanes 16-31 ->
// (M = base+8+i). A row's 16 cols live in one 16-lane half -> shfl_xor reduce.
__device__ __forceinline__ void wave_stats(
    const v8f acc[2][2], int tid, int row0, int chunk,
    float* sbm, float* sbs,                 // LDS [MT][4]
    float* __restrict__ gmax, float* __restrict__ gsum)
{
  const int lane = tid & 31;
  const int wave = tid >> 5;
  const int wm = wave >> 2, wn = wave & 3;
  const int m = lane & 15, half = lane >> 4;

  #pragma unroll
  for (int r = 0; r < 2; ++r) {
    #pragma unroll
    for (int i = 0; i < 8; ++i) {
      float v0 = acc[r][0][i], v1 = acc[r][1][i];
      float mx = fmaxf(v0, v1);
      #pragma unroll
      for (int d = 1; d < 16; d <<= 1) mx = fmaxf(mx, __shfl_xor(mx, d, 32));
      float s = __expf(v0 - mx) + __expf(v1 - mx);
      #pragma unroll
      for (int d = 1; d < 16; d <<= 1) s += __shfl_xor(s, d, 32);
      if (m == 0) {
        int row = wm * 32 + r * 16 + half * 8 + i;
        sbm[row * 4 + wn] = mx;
        sbs[row * 4 + wn] = s;
      }
    }
  }
  __syncthreads();
  if (tid < MT) {
    float M = -3.0e38f;
    #pragma unroll
    for (int j = 0; j < 4; ++j) M = fmaxf(M, sbm[tid * 4 + j]);
    float S = 0.0f;
    #pragma unroll
    for (int j = 0; j < 4; ++j) S += sbs[tid * 4 + j] * __expf(sbm[tid * 4 + j] - M);
    gmax[(size_t)chunk * BT_ + row0 + tid] = M;
    gsum[(size_t)chunk * BT_ + row0 + tid] = S;
  }
  __syncthreads();
}

__global__ __launch_bounds__(256) void jsd_pass1(
    const float* __restrict__ Xs, const float* __restrict__ Ws,
    const float* __restrict__ Xt, const float* __restrict__ Wt,
    float* smax, float* ssum, float* tmax, float* tsum)
{
  __shared__ stage_t lA[2][AE];
  __shared__ stage_t lB[2][BE];
  __shared__ float   sbm[MT * 4];
  __shared__ float   sbs[MT * 4];
  const int tid   = threadIdx.x;
  const int row0  = blockIdx.x * MT;   // row tile varies fastest -> L2 reuse
  const int col0  = blockIdx.y * NT;
  const int chunk = blockIdx.y;

  v8f acc[2][2];
  gemm64(Xs, Ws, HS_, row0, col0, &lA[0][0], &lB[0][0], tid, acc);
  wave_stats(acc, tid, row0, chunk, sbm, sbs, smax, ssum);

  gemm64(Xt, Wt, HT_, row0, col0, &lA[0][0], &lB[0][0], tid, acc);
  wave_stats(acc, tid, row0, chunk, sbm, sbs, tmax, tsum);
}

__global__ void jsd_combine(const float* __restrict__ smax, const float* __restrict__ ssum,
                            const float* __restrict__ tmax, const float* __restrict__ tsum,
                            float* Ls, float* Lt)
{
  int row = blockIdx.x * blockDim.x + threadIdx.x;
  if (row >= BT_) return;
  float M = -3.0e38f;
  for (int j = 0; j < NCH; ++j) M = fmaxf(M, smax[(size_t)j * BT_ + row]);
  float S = 0.0f;
  for (int j = 0; j < NCH; ++j)
    S += ssum[(size_t)j * BT_ + row] * __expf(smax[(size_t)j * BT_ + row] - M);
  Ls[row] = M + __logf(S);

  M = -3.0e38f;
  for (int j = 0; j < NCH; ++j) M = fmaxf(M, tmax[(size_t)j * BT_ + row]);
  S = 0.0f;
  for (int j = 0; j < NCH; ++j)
    S += tsum[(size_t)j * BT_ + row] * __expf(tmax[(size_t)j * BT_ + row] - M);
  Lt[row] = M + __logf(S);
}

__global__ __launch_bounds__(256) void jsd_pass2(
    const float* __restrict__ Xs, const float* __restrict__ Ws,
    const float* __restrict__ Xt, const float* __restrict__ Wt,
    const int* __restrict__ target,
    const float* __restrict__ Ls, const float* __restrict__ Lt,
    float* partials)
{
  __shared__ stage_t lA[2][AE];
  __shared__ stage_t lB[2][BE];
  __shared__ float   lLs[MT];
  __shared__ float   lLt[MT];
  __shared__ int     lTg[MT];
  __shared__ float   red[256];
  const int tid  = threadIdx.x;
  const int row0 = blockIdx.x * MT;
  const int col0 = blockIdx.y * NT;
  const int lane = tid & 31;
  const int wave = tid >> 5;
  const int wm = wave >> 2, wn = wave & 3;
  const int m = lane & 15, half = lane >> 4;

  if (tid < MT) {
    lLs[tid] = Ls[row0 + tid];
    lLt[tid] = Lt[row0 + tid];
    lTg[tid] = target[row0 + tid];
  }
  // (gemm64's internal barriers order these LDS writes before their reads)

  v8f accS[2][2], accT[2][2];
  gemm64(Xs, Ws, HS_, row0, col0, &lA[0][0], &lB[0][0], tid, accS);
  gemm64(Xt, Wt, HT_, row0, col0, &lA[0][0], &lB[0][0], tid, accT);

  // contribution = 0.25*(ps*(lps-logm) + pt*(lpt-logm)) [soft, beta=0.5]
  //              + 0.5*(-lps) at the target column        [hard CE]
  float loc = 0.0f;
  #pragma unroll
  for (int r = 0; r < 2; ++r) {
    #pragma unroll
    for (int c = 0; c < 2; ++c) {
      const int col = col0 + wn * 32 + c * 16 + m;
      #pragma unroll
      for (int i = 0; i < 8; ++i) {
        const int row = wm * 32 + r * 16 + half * 8 + i;
        float lps = accS[r][c][i] - lLs[row];
        float lpt = accT[r][c][i] - lLt[row];
        float mx  = fmaxf(lps, lpt);
        float logm = mx + __logf(0.5f * (__expf(lps - mx) + __expf(lpt - mx)));
        loc += 0.25f * (__expf(lps) * (lps - logm) + __expf(lpt) * (lpt - logm));
        int tgt = lTg[row];
        if (tgt != -100 && col == tgt) loc -= 0.5f * lps;
      }
    }
  }
  red[tid] = loc;
  __syncthreads();
  for (int st = 128; st > 0; st >>= 1) {
    if (tid < st) red[tid] += red[tid + st];
    __syncthreads();
  }
  if (tid == 0) partials[blockIdx.y * gridDim.x + blockIdx.x] = red[0];
}

__global__ void jsd_reduce(const float* __restrict__ partials, int n, float* out)
{
  __shared__ float red[256];
  float s = 0.0f;
  for (int i = threadIdx.x; i < n; i += 256) s += partials[i];
  red[threadIdx.x] = s;
  __syncthreads();
  for (int st = 128; st > 0; st >>= 1) {
    if (threadIdx.x < st) red[threadIdx.x] += red[threadIdx.x + st];
    __syncthreads();
  }
  if (threadIdx.x == 0) out[0] = red[0] * (1.0f / (float)BT_);
}

extern "C" void kernel_launch(void* const* d_in, const int* in_sizes, int n_in,
                              void* d_out, int out_size, void* d_ws, size_t ws_size,
                              hipStream_t stream)
{
  (void)in_sizes; (void)n_in; (void)out_size; (void)ws_size;
  const float* Xs = (const float*)d_in[0];   // student_input  (2048 x 2048)
  const float* Xt = (const float*)d_in[1];   // teacher_input  (2048 x 4096)
  const float* Ws = (const float*)d_in[2];   // student_weight (32000 x 2048)
  const float* Wt = (const float*)d_in[3];   // teacher_weight (32000 x 4096)
  const int*   tg = (const int*)d_in[4];     // target (2048)

  float* ws   = (float*)d_ws;
  float* smax = ws;
  float* ssum = smax + (size_t)NCH * BT_;
  float* tmax = ssum + (size_t)NCH * BT_;
  float* tsum = tmax + (size_t)NCH * BT_;
  float* Ls   = tsum + (size_t)NCH * BT_;
  float* Lt   = Ls + BT_;
  float* partials = Lt + BT_;                // NCH * NRT = 8000 floats

  dim3 grid(NRT, NCH);                       // x = row tile (fast) for L2 reuse
  dim3 block(256);
  jsd_pass1<<<grid, block, 0, stream>>>(Xs, Ws, Xt, Wt, smax, ssum, tmax, tsum);
  jsd_combine<<<(BT_ + 255) / 256, 256, 0, stream>>>(smax, ssum, tmax, tsum, Ls, Lt);
  jsd_pass2<<<grid, block, 0, stream>>>(Xs, Ws, Xt, Wt, tg, Ls, Lt, partials);
  jsd_reduce<<<1, 256, 0, stream>>>(partials, NCH * NRT, (float*)d_out);
}
